// BiLSTM_crf_14164802142660
// MI455X (gfx1250) — matrix operand order repaired and checked by reading, not compiled, not run on hardware
//
#include <hip/hip_runtime.h>
#include <hip/hip_bf16.h>
#include <math.h>

#define DEVFN __device__ __forceinline__

typedef __bf16 bf16;
typedef __attribute__((ext_vector_type(16))) __bf16 v16bf;
typedef __attribute__((ext_vector_type(8)))  __bf16 v8bf;
typedef __attribute__((ext_vector_type(4)))  __bf16 v4bf;
typedef __attribute__((ext_vector_type(8)))  float  v8f;
typedef __attribute__((ext_vector_type(4)))  unsigned int u32x4;
typedef __attribute__((ext_vector_type(8)))  int i32x8;
typedef __attribute__((ext_vector_type(4)))  int i32x4;

// Problem constants (match reference)
constexpr int VOCAB = 32000;
constexpr int B  = 64;
constexpr int T  = 512;
constexpr int E  = 256;
constexpr int H  = 256;
constexpr int NK = 10;
constexpr int G4H = 4 * H;   // 1024 gate columns

// ---------------- workspace layout (bytes) ----------------
constexpr size_t OFF_XBF  = 0;                               // [T*B, E] bf16 (rows = t*B+b)
constexpr size_t SZ_XBF   = (size_t)T * B * E * 2;
constexpr size_t SZ_W     = (size_t)G4H * H * 2;             // 1024x256 bf16
constexpr size_t OFF_WIHF = OFF_XBF  + SZ_XBF;
constexpr size_t OFF_WIHB = OFF_WIHF + SZ_W;
constexpr size_t OFF_WHHF = OFF_WIHB + SZ_W;
constexpr size_t OFF_WHHB = OFF_WHHF + SZ_W;
constexpr size_t SZ_GX    = (size_t)T * B * G4H * 4;         // f32
constexpr size_t OFF_GXF  = OFF_WHHB + SZ_W;
constexpr size_t OFF_GXB  = OFF_GXF + SZ_GX;
constexpr size_t OFF_GSCR = OFF_GXB + SZ_GX;                 // [2][B*4H] f32 gate scratch
constexpr size_t SZ_GSCR  = (size_t)2 * B * G4H * 4;
constexpr size_t OFF_CST  = OFF_GSCR + SZ_GSCR;              // [2][B*H] f32 cell state
constexpr size_t SZ_CST   = (size_t)2 * B * H * 4;
constexpr size_t SZ_HS    = (size_t)T * B * H * 4;           // f32
constexpr size_t OFF_HSF  = OFF_CST + SZ_CST;
constexpr size_t OFF_HSB  = OFF_HSF + SZ_HS;
constexpr size_t OFF_LOG  = OFF_HSB + SZ_HS;                 // [T*B, NK] f32
constexpr size_t SZ_LOG   = (size_t)T * B * NK * 4;
constexpr size_t OFF_BP   = OFF_LOG + SZ_LOG;                // [(T-1)*B, NK] i32 backptrs

// ---------------- fragment helpers (CDNA5 WMMA layouts) ----------------
// A fragment: 16x32 (MxK) bf16. lane<16: K = {0..7, 16..23}; lane>=16: K+8.
DEVFN v16bf load_a_frag(const bf16* base, int ld) {
  const int lane = threadIdx.x & 31;
  const int m    = lane & 15;
  const int kb   = (lane < 16) ? 0 : 8;
  const bf16* p  = base + (size_t)m * ld + kb;
  v8bf lo = *(const v8bf*)p;         // K = kb + 0..7
  v8bf hi = *(const v8bf*)(p + 16);  // K = kb + 16..23
  v16bf r;
#pragma unroll
  for (int j = 0; j < 8; ++j) { r[j] = lo[j]; r[8 + j] = hi[j]; }
  return r;
}

// B fragment: 32x16 (KxN) where B[k][n] = W[n*ld + k] (row-major weight acts as B^T).
// lane<16: K = {0..7, 16..23}, lane>=16: K+8; N = n0 + (lane&15).
DEVFN v16bf load_b_fragT(const bf16* w, int ld, int n0, int k0) {
  const int lane = threadIdx.x & 31;
  const bf16* p  = w + (size_t)(n0 + (lane & 15)) * ld + k0 + ((lane < 16) ? 0 : 8);
  v8bf lo = *(const v8bf*)p;
  v8bf hi = *(const v8bf*)(p + 16);
  v16bf r;
#pragma unroll
  for (int j = 0; j < 8; ++j) { r[j] = lo[j]; r[8 + j] = hi[j]; }
  return r;
}

DEVFN float sigm(float x) { return 1.0f / (1.0f + __expf(-x)); }

// ---------------- kernels ----------------
// f32 -> bf16, 4 elements/thread (b128 load, b64 store)
__global__ void cvt_bf16_kernel(const float* __restrict__ src, bf16* __restrict__ dst, int n4) {
  int i = blockIdx.x * blockDim.x + threadIdx.x;
  if (i >= n4) return;
  const float4 v = ((const float4*)src)[i];
  v4bf o; o[0] = (bf16)v.x; o[1] = (bf16)v.y; o[2] = (bf16)v.z; o[3] = (bf16)v.w;
  ((v4bf*)dst)[i] = o;
}

// x[t*B+b][e..e+3] = emb[sent[b*T+t]][e..e+3], converted to bf16
__global__ void embed_kernel(const int* __restrict__ sent, const float* __restrict__ emb,
                             bf16* __restrict__ xbf) {
  size_t idx4 = ((size_t)blockIdx.x * blockDim.x + threadIdx.x) * 4;   // < T*B*E
  int e   = (int)(idx4 & (E - 1));
  int row = (int)(idx4 >> 8);      // t*B + b
  int t   = row >> 6;
  int b   = row & 63;
  int tok = sent[b * T + t];
  const float4 v = *(const float4*)(emb + (size_t)tok * E + e);
  v4bf o; o[0] = (bf16)v.x; o[1] = (bf16)v.y; o[2] = (bf16)v.z; o[3] = (bf16)v.w;
  *(v4bf*)(xbf + idx4) = o;
}

// Gx = X @ Wih^T + bias  (M=T*B=32768, K=256, N=1024), bf16 WMMA, f32 out.
// block = 256 threads (8 waves); wave computes 16(M) x 64(N); grid (16, 256, 2).
__global__ void gx_gemm_kernel(const bf16* __restrict__ xbf,
                               const bf16* __restrict__ wih_f, const bf16* __restrict__ wih_b,
                               const float* __restrict__ bias_f, const float* __restrict__ bias_b,
                               float* __restrict__ gx_f, float* __restrict__ gx_b) {
  const int dir = blockIdx.z;
  const bf16*  wih  = dir ? wih_b  : wih_f;
  const float* bias = dir ? bias_b : bias_f;
  float*       gx   = dir ? gx_b   : gx_f;

  const int wave  = threadIdx.x >> 5;
  const int lane  = threadIdx.x & 31;
  const int l15   = lane & 15;
  const int mbase = (lane < 16) ? 0 : 8;
  const int mrow0 = blockIdx.y * 128 + wave * 16;
  const int ncol0 = blockIdx.x * 64;

  v8f acc[4];
#pragma unroll
  for (int nb = 0; nb < 4; ++nb) {
    float bv = bias[ncol0 + nb * 16 + l15];
#pragma unroll
    for (int q = 0; q < 8; ++q) acc[nb][q] = bv;
  }
  // software-pipelined over K: prefetch A(kt+1) while issuing WMMAs for kt
  v16bf a_cur = load_a_frag(xbf + (size_t)mrow0 * E, E);
#pragma unroll
  for (int kt = 0; kt < 8; ++kt) {
    v16bf a_nxt;
    if (kt < 7) a_nxt = load_a_frag(xbf + (size_t)mrow0 * E + (kt + 1) * 32, E);
#pragma unroll
    for (int nb = 0; nb < 4; ++nb) {
      v16bf bf = load_b_fragT(wih, E, ncol0 + nb * 16, kt * 32);
      acc[nb] = __builtin_amdgcn_wmma_f32_16x16x32_bf16(false, a_cur, false, bf, (short)0,
                                                        acc[nb], false, false);
    }
    a_cur = a_nxt;
  }
#pragma unroll
  for (int nb = 0; nb < 4; ++nb)
#pragma unroll
    for (int q = 0; q < 8; ++q)
      gx[(size_t)(mrow0 + mbase + q) * G4H + ncol0 + nb * 16 + l15] = acc[nb][q];
}

// Persistent per-direction LSTM: grid(2), block(512) = 16 waves.
// Per step: g = Gx[t] + h @ Whh^T via WMMA (h kept as bf16 in LDS), then gate math.
__global__ void lstm_kernel(const float* __restrict__ gx_f, const float* __restrict__ gx_b,
                            const bf16* __restrict__ whh_f, const bf16* __restrict__ whh_b,
                            float* __restrict__ g_scr,   // [2][B*4H]
                            float* __restrict__ c_scr,   // [2][B*H]
                            float* __restrict__ hs_f, float* __restrict__ hs_b) {
  const int dir = blockIdx.x;
  const float* gx  = dir ? gx_b  : gx_f;
  const bf16*  whh = dir ? whh_b : whh_f;
  float* g  = g_scr + (size_t)dir * B * G4H;
  float* cs = c_scr + (size_t)dir * B * H;
  float* hs = dir ? hs_b : hs_f;

  __shared__ bf16 h_lds[B * H];   // 32 KB: h_prev as WMMA A-source

  const int tid   = threadIdx.x;
  const int wave  = tid >> 5;
  const int lane  = tid & 31;
  const int l15   = lane & 15;
  const int mbase = (lane < 16) ? 0 : 8;

  for (int i = tid; i < B * H; i += blockDim.x) { h_lds[i] = (bf16)0.0f; cs[i] = 0.0f; }
  __syncthreads();

  for (int s = 0; s < T; ++s) {
    const int t = dir ? (T - 1 - s) : s;
    const float* gxt = gx + (size_t)t * B * G4H;

    // prefetch next step's Gx slice (256 KB) toward L0/L2 while this step computes
    if (s + 1 < T) {
      const int tn = dir ? (T - 2 - s) : (s + 1);
      const char* nx = (const char*)(gx + (size_t)tn * B * G4H);
#pragma unroll
      for (int r = 0; r < 4; ++r)
        __builtin_prefetch(nx + ((size_t)tid + (size_t)r * 512) * 128, 0, 3);
    }

    // ---- recurrent GEMM: 4 mtiles x 64 ntiles split over 16 waves (4 ntiles each)
    for (int nt = 0; nt < 4; ++nt) {
      const int n0 = (wave * 4 + nt) * 16;
      const int n  = n0 + l15;
      v8f acc[4];
#pragma unroll
      for (int mt = 0; mt < 4; ++mt)
#pragma unroll
        for (int q = 0; q < 8; ++q)
          acc[mt][q] = gxt[(size_t)(mt * 16 + mbase + q) * G4H + n];
      // double-buffer the Whh B-fragment across the K loop
      v16bf b_cur = load_b_fragT(whh, H, n0, 0);
#pragma unroll
      for (int kt = 0; kt < 8; ++kt) {
        v16bf b_nxt;
        if (kt < 7) b_nxt = load_b_fragT(whh, H, n0, (kt + 1) * 32);
#pragma unroll
        for (int mt = 0; mt < 4; ++mt) {
          v16bf a = load_a_frag(h_lds + mt * 16 * H + kt * 32, H);
          acc[mt] = __builtin_amdgcn_wmma_f32_16x16x32_bf16(false, a, false, b_cur, (short)0,
                                                            acc[mt], false, false);
        }
        b_cur = b_nxt;
      }
#pragma unroll
      for (int mt = 0; mt < 4; ++mt)
#pragma unroll
        for (int q = 0; q < 8; ++q)
          g[(size_t)(mt * 16 + mbase + q) * G4H + n] = acc[mt][q];
    }
    __threadfence_block();
    __syncthreads();

    // ---- activations: c = sig(f)*c + sig(i)*tanh(g); h = sig(o)*tanh(c)
    for (int idx = tid; idx < B * H; idx += blockDim.x) {
      const int b = idx >> 8;
      const int j = idx & (H - 1);
      const float* gb = g + (size_t)b * G4H;
      float gi = gb[j], gf = gb[H + j], gg = gb[2 * H + j], go = gb[3 * H + j];
      float cc = sigm(gf) * cs[idx] + sigm(gi) * tanhf(gg);
      cs[idx] = cc;
      float hh = sigm(go) * tanhf(cc);
      h_lds[b * H + j] = (bf16)hh;
      hs[((size_t)t * B + b) * H + j] = hh;
    }
    __threadfence_block();
    __syncthreads();
  }
}

// One thread per (t,b): 10 accumulators; W_out staged into LDS by the
// Tensor Data Mover (TENSOR_LOAD_TO_LDS), waited with s_wait_tensorcnt.
__global__ void logits_kernel(const float* __restrict__ hs_f, const float* __restrict__ hs_b,
                              const float* __restrict__ wout, const float* __restrict__ bout,
                              float* __restrict__ logit) {
  __shared__ float wl[NK * 2 * H];   // 20 KB = 5120 f32, contiguous tile
  __shared__ float bl[NK];
  if (threadIdx.x < NK) bl[threadIdx.x] = bout[threadIdx.x];

  // ---- TDM: DMA W_out (1 x 5120 f32 tile) from global into LDS (wave 0 issues)
  if (threadIdx.x < 32) {
    const unsigned long long ga = (unsigned long long)(uintptr_t)wout;
    const unsigned int lds_addr = (unsigned int)(uintptr_t)wl;  // LDS aperture: low 32b = LDS byte offset
    const unsigned int NEL = NK * 2 * H;                        // 5120 elements of 4 bytes
    // D# group0: [1:0]=count=1 | [63:32]=lds_addr | [120:64]=global_addr | [127:126]=type=2
    u32x4 g0;
    g0[0] = 1u;                                   // count=1, user descriptor
    g0[1] = lds_addr;
    g0[2] = (unsigned int)(ga & 0xFFFFFFFFu);
    g0[3] = ((unsigned int)(ga >> 32) & 0x01FFFFFFu) | (2u << 30);
    // D# group1: data_size=2 (4B); tensor_dim0=tile_dim0=NEL; tensor_dim1=tile_dim1=1;
    //            tensor_dim0_stride=NEL; no padding/iterate/multicast.
    i32x8 g1;
    g1[0] = (int)(2u << 16);                      // wg_mask=0, data_size=4B
    g1[1] = (int)((NEL & 0xFFFFu) << 16);         // atomic_barrier_addr=0 | tensor_dim0[15:0]
    g1[2] = (int)((NEL >> 16) | (1u << 16));      // tensor_dim0[31:16] | tensor_dim1[15:0]=1
    g1[3] = (int)((NEL & 0xFFFFu) << 16);         // tensor_dim1[31:16]=0 | tile_dim0=NEL
    g1[4] = (int)1u;                              // tile_dim1=1 | tile_dim2=0
    g1[5] = (int)NEL;                             // tensor_dim0_stride[31:0]
    g1[6] = 0;                                    // stride0[47:32]=0 | stride1[15:0]=0
    g1[7] = 0;
    i32x4 z4 = {0, 0, 0, 0};                      // groups 2/3 unused (<=2D tensor)
    i32x8 z8 = {0, 0, 0, 0, 0, 0, 0, 0};          // extra group (clang-23 6-arg form)
    __builtin_amdgcn_tensor_load_to_lds(g0, g1, z4, z4, z8, 0);
    __builtin_amdgcn_s_wait_tensorcnt(0);
  }
  __syncthreads();

  const int tb = blockIdx.x * blockDim.x + threadIdx.x;   // t*B + b
  if (tb >= T * B) return;
  const float* hf = hs_f + (size_t)tb * H;
  const float* hb = hs_b + (size_t)tb * H;
  float acc[NK];
#pragma unroll
  for (int k = 0; k < NK; ++k) acc[k] = bl[k];
  for (int i = 0; i < H; i += 4) {
    const float4 v = *(const float4*)(hf + i);
#pragma unroll
    for (int k = 0; k < NK; ++k) {
      const float* w = wl + k * 2 * H + i;
      acc[k] += v.x * w[0] + v.y * w[1] + v.z * w[2] + v.w * w[3];
    }
  }
  for (int i = 0; i < H; i += 4) {
    const float4 v = *(const float4*)(hb + i);
#pragma unroll
    for (int k = 0; k < NK; ++k) {
      const float* w = wl + k * 2 * H + H + i;
      acc[k] += v.x * w[0] + v.y * w[1] + v.z * w[2] + v.w * w[3];
    }
  }
#pragma unroll
  for (int k = 0; k < NK; ++k) logit[(size_t)tb * NK + k] = acc[k];
}

// Batched Viterbi: one thread per batch element. out = [scores(B) | path(B*T) as float]
__global__ void viterbi_kernel(const float* __restrict__ logit, const float* __restrict__ trans,
                               int* __restrict__ bp, float* __restrict__ out) {
  __shared__ float tr[NK * NK];
  const int tid = threadIdx.x;
  for (int i = tid; i < NK * NK; i += blockDim.x) tr[i] = trans[i];
  __syncthreads();
  if (tid >= B) return;
  const int b = tid;
  float v[NK];
#pragma unroll
  for (int k = 0; k < NK; ++k) v[k] = logit[(size_t)b * NK + k];
  for (int t = 1; t < T; ++t) {
    const float* lt = logit + (size_t)(t * B + b) * NK;
    float nv[NK];
#pragma unroll
    for (int j = 0; j < NK; ++j) {
      float best = v[0] + tr[j];
      int bi = 0;
#pragma unroll
      for (int i = 1; i < NK; ++i) {
        float c = v[i] + tr[i * NK + j];
        if (c > best) { best = c; bi = i; }
      }
      nv[j] = best + lt[j];
      bp[((size_t)(t - 1) * B + b) * NK + j] = bi;
    }
#pragma unroll
    for (int j = 0; j < NK; ++j) v[j] = nv[j];
  }
  float best = v[0];
  int last = 0;
#pragma unroll
  for (int k = 1; k < NK; ++k) if (v[k] > best) { best = v[k]; last = k; }
  out[b] = best;
  float* path = out + B;
  path[(size_t)b * T + (T - 1)] = (float)last;
  for (int t = T - 2; t >= 0; --t) {
    last = bp[((size_t)t * B + b) * NK + last];
    path[(size_t)b * T + t] = (float)last;
  }
}

// ---------------- host launcher ----------------
extern "C" void kernel_launch(void* const* d_in, const int* in_sizes, int n_in,
                              void* d_out, int out_size, void* d_ws, size_t ws_size,
                              hipStream_t stream) {
  (void)in_sizes; (void)n_in; (void)out_size; (void)ws_size;
  const int*   sent  = (const int*)d_in[0];
  // d_in[1] = lengths (all == T in reference; unused)
  const float* emb   = (const float*)d_in[2];
  const float* Wih_f = (const float*)d_in[3];
  const float* Whh_f = (const float*)d_in[4];
  const float* b_f   = (const float*)d_in[5];
  const float* Wih_b = (const float*)d_in[6];
  const float* Whh_b = (const float*)d_in[7];
  const float* b_b   = (const float*)d_in[8];
  const float* W_out = (const float*)d_in[9];
  const float* b_out = (const float*)d_in[10];
  const float* trans = (const float*)d_in[11];

  char* ws = (char*)d_ws;
  bf16*  xbf    = (bf16*)(ws + OFF_XBF);
  bf16*  wihf16 = (bf16*)(ws + OFF_WIHF);
  bf16*  wihb16 = (bf16*)(ws + OFF_WIHB);
  bf16*  whhf16 = (bf16*)(ws + OFF_WHHF);
  bf16*  whhb16 = (bf16*)(ws + OFF_WHHB);
  float* gx_f   = (float*)(ws + OFF_GXF);
  float* gx_b   = (float*)(ws + OFF_GXB);
  float* g_scr  = (float*)(ws + OFF_GSCR);
  float* c_scr  = (float*)(ws + OFF_CST);
  float* hs_f   = (float*)(ws + OFF_HSF);
  float* hs_b   = (float*)(ws + OFF_HSB);
  float* logit  = (float*)(ws + OFF_LOG);
  int*   bp     = (int*)(ws + OFF_BP);

  const int NW4 = (G4H * H) / 4;  // 65536 float4 groups per weight matrix (E == H)
  cvt_bf16_kernel<<<NW4 / 256, 256, 0, stream>>>(Wih_f, wihf16, NW4);
  cvt_bf16_kernel<<<NW4 / 256, 256, 0, stream>>>(Wih_b, wihb16, NW4);
  cvt_bf16_kernel<<<NW4 / 256, 256, 0, stream>>>(Whh_f, whhf16, NW4);
  cvt_bf16_kernel<<<NW4 / 256, 256, 0, stream>>>(Whh_b, whhb16, NW4);

  embed_kernel<<<(T * B * E / 4) / 256, 256, 0, stream>>>(sent, emb, xbf);

  dim3 gg(G4H / 64, (T * B) / 128, 2);
  gx_gemm_kernel<<<gg, 256, 0, stream>>>(xbf, wihf16, wihb16, b_f, b_b, gx_f, gx_b);

  lstm_kernel<<<2, 512, 0, stream>>>(gx_f, gx_b, whhf16, whhb16, g_scr, c_scr, hs_f, hs_b);

  logits_kernel<<<(T * B + 255) / 256, 256, 0, stream>>>(hs_f, hs_b, W_out, b_out, logit);

  viterbi_kernel<<<1, 64, 0, stream>>>(logit, trans, bp, (float*)d_out);
}
